// Geo_49203145343434
// MI455X (gfx1250) — compile-verified, matrix operand order checked by reading, not run on hardware
//
#include <hip/hip_runtime.h>
#include <hip/hip_bf16.h>

// ---------------------------------------------------------------------------
// Problem constants (from reference)
// ---------------------------------------------------------------------------
#define DIM_IN   128
#define DIM_Z    258
#define KPAD     288            // DIM_Z padded up to multiple of 32 for WMMA
#define DIM_MED  512
#define DIM_OUT  2
#define N1       3072
#define N2       3072
#define NTOT     6144           // N1 + N2
#define INV_TEMP 0.062257280636f  // 1/sqrt(258)  (folded into Qh)
#define INV_BERN 20.0f            // 1/0.05
#define LOGIT_MAX 6.9067547786f   // log(0.999/0.001) -- clip(sigmoid) in logit space

#define LOG2E 1.44269504089f
#define LN2   0.69314718056f

typedef _Float16 half16 __attribute__((ext_vector_type(16)));
typedef _Float16 half8  __attribute__((ext_vector_type(8)));
typedef float    floatx8 __attribute__((ext_vector_type(8)));

union HalfVec { half16 v; half8 h[2]; };

// raw-hardware transcendentals (v_exp_f32 / v_log_f32 / v_rcp_f32, no guards)
__device__ __forceinline__ float fast_exp(float x) {
    return __builtin_amdgcn_exp2f(x * LOG2E);
}
__device__ __forceinline__ float fast_log(float x) {
    return __builtin_amdgcn_logf(x) * LN2;
}
__device__ __forceinline__ float fast_sigmoid(float x) {
    return __builtin_amdgcn_rcpf(1.0f + __builtin_amdgcn_exp2f(-x * LOG2E));
}

// ---------------------------------------------------------------------------
// Kernel 1: Q/K projections -> f16 [NTOT][KPAD]; Q pre-scaled by 1/sqrt(258).
// one block per row, 256 threads
// ---------------------------------------------------------------------------
__global__ void geo_proj_kernel(const float* __restrict__ lm_X,
                                const float* __restrict__ tg_X,
                                const float* __restrict__ Wq,
                                const float* __restrict__ bq,
                                const float* __restrict__ Wk,
                                const float* __restrict__ bk,
                                _Float16* __restrict__ Qh,
                                _Float16* __restrict__ Kh) {
    const int row = blockIdx.x;
    const int tid = threadIdx.x;
    const float* __restrict__ x =
        (row < N1) ? (lm_X + (size_t)row * DIM_IN)
                   : (tg_X + (size_t)(row - N1) * DIM_IN);

    __shared__ float xs[DIM_IN];
    if (tid < DIM_IN) xs[tid] = x[tid];
    __syncthreads();

    for (int c = tid; c < KPAD; c += 256) {
        float aq = 0.0f, ak = 0.0f;
        if (c < DIM_Z) {
            aq = bq[c];
            ak = bk[c];
            #pragma unroll 4
            for (int i = 0; i < DIM_IN; ++i) {
                const float xi = xs[i];
                aq = fmaf(xi, Wq[(size_t)i * DIM_Z + c], aq);
                ak = fmaf(xi, Wk[(size_t)i * DIM_Z + c], ak);
            }
            aq *= INV_TEMP;   // fold temperature into Q
        }
        Qh[(size_t)row * KPAD + c] = (_Float16)aq;
        Kh[(size_t)row * KPAD + c] = (_Float16)ak;
    }
}

// ---------------------------------------------------------------------------
// Kernel 2: fused score GEMM (WMMA f32 <- f16) + elementwise epilogue.
//   S = (Q/sqrt(258)) @ K^T                         (scale pre-folded)
//   prob = clip(sigmoid(S), 1e-3, 0.999)            -> out_prob
//   logit(prob) == clamp(S, +-LOGIT_MAX)            (exact identity)
//   mask = sigmoid((logit+logistic(u)) * 20)        -> out_mask
//   adj  = exp(S) * mask  for rows>=N1, cols<N1     -> ws adj block
// 128x64 tile per workgroup; 4 waves, each wave: 32 rows x 64 cols
// -> 8 WMMA per 12 b128 loads per K-chunk.
// ---------------------------------------------------------------------------
__global__ void __launch_bounds__(128)
geo_score_kernel(const _Float16* __restrict__ Qh,
                 const _Float16* __restrict__ Kh,
                 const float* __restrict__ noise_u,
                 float* __restrict__ out_prob,
                 float* __restrict__ out_mask,
                 float* __restrict__ adj) {
    const int wid  = threadIdx.x >> 5;       // wave id 0..3
    const int lane = threadIdx.x & 31;
    const int lr   = lane & 15;              // row/col within 16-tile
    const int lh   = lane >> 4;              // half-wave selector

    const int rowBase  = blockIdx.y * 128 + wid * 32;
    const int colBase0 = blockIdx.x * 64;

    floatx8 acc[2][4];
    #pragma unroll
    for (int r = 0; r < 2; ++r)
        #pragma unroll
        for (int ct = 0; ct < 4; ++ct)
            acc[r][ct] = (floatx8){0,0,0,0,0,0,0,0};

    // A fragment source rows: (rowBase + r*16 + lr), k offsets {32kc+8lh..} & +16
    const _Float16* __restrict__ qrow0 = Qh + (size_t)(rowBase + lr) * KPAD;
    const _Float16* __restrict__ qrow1 = qrow0 + (size_t)16 * KPAD;

    #pragma unroll 1
    for (int kc = 0; kc < KPAD / 32; ++kc) {
        const int kb = kc * 32 + lh * 8;

        HalfVec A0, A1;
        A0.h[0] = *(const half8*)(qrow0 + kb);
        A0.h[1] = *(const half8*)(qrow0 + kb + 16);
        A1.h[0] = *(const half8*)(qrow1 + kb);
        A1.h[1] = *(const half8*)(qrow1 + kb + 16);

        #pragma unroll
        for (int ct = 0; ct < 4; ++ct) {
            const _Float16* __restrict__ krow =
                Kh + (size_t)(colBase0 + ct * 16 + lr) * KPAD;
            HalfVec B;
            B.h[0] = *(const half8*)(krow + kb);
            B.h[1] = *(const half8*)(krow + kb + 16);

            acc[0][ct] = __builtin_amdgcn_wmma_f32_16x16x32_f16(
                false, A0.v, false, B.v, (short)0, acc[0][ct], false, false);
            acc[1][ct] = __builtin_amdgcn_wmma_f32_16x16x32_f16(
                false, A1.v, false, B.v, (short)0, acc[1][ct], false, false);
        }
    }

    // Epilogue. D layout: lane holds N = lane&15, M = v + 8*(lane>>4), v=0..7.
    #pragma unroll
    for (int r = 0; r < 2; ++r) {
        #pragma unroll
        for (int ct = 0; ct < 4; ++ct) {
            const int col = colBase0 + ct * 16 + lr;
            #pragma unroll
            for (int v = 0; v < 8; ++v) {
                const int row = rowBase + r * 16 + lh * 8 + v;
                const size_t idx = (size_t)row * NTOT + col;

                const float s = acc[r][ct][v];

                float p = fast_sigmoid(s);
                p = fminf(fmaxf(p, 0.001f), 0.999f);
                out_prob[idx] = p;

                // logit(clip(sigmoid(s))) == clamp(s, +-LOGIT_MAX), exactly
                const float logits = fminf(fmaxf(s, -LOGIT_MAX), LOGIT_MAX);
                const float u = noise_u[idx];
                const float logistic = fast_log(u) - fast_log(1.0f - u);
                const float m = fast_sigmoid((logits + logistic) * INV_BERN);
                out_mask[idx] = m;

                if (row >= N1 && col < N1) {
                    adj[(size_t)(row - N1) * N1 + col] = fast_exp(s) * m;
                }
            }
        }
    }
}

// ---------------------------------------------------------------------------
// Kernel 3: per-tg-row fused tail.
//   deg      = sum_j adj[row][j] + 1
//   agg[d]   = (sum_j adj[row][j]*lm_feat[j][d] + tg_feat0[row][d]) / deg
//   f1       = agg @ W1 + b1                       (130)
//   fin      = [tg_X[row], f1]                     (258)
//   h        = relu(fin @ Wp1 + bp1)               (512)
//   y        = h @ Wp2 + bp2                       (2)
// one block (256 threads) per tg row.
// ---------------------------------------------------------------------------
#define CHUNK 512
#define DF    (DIM_IN + 2)   // 130

__global__ void __launch_bounds__(256)
geo_tail_kernel(const float* __restrict__ adj,
                const float* __restrict__ lm_X,
                const float* __restrict__ lm_Y,
                const float* __restrict__ tg_X,
                const float* __restrict__ W1,
                const float* __restrict__ b1,
                const float* __restrict__ Wp1,
                const float* __restrict__ bp1,
                const float* __restrict__ Wp2,
                const float* __restrict__ bp2,
                float* __restrict__ y_out) {
    const int row = blockIdx.x;
    const int tid = threadIdx.x;

    __shared__ float lds[CHUNK];
    __shared__ float red[256];
    __shared__ float sAgg[DF];
    __shared__ float sFin[DIM_Z];

    const float* __restrict__ arow = adj + (size_t)row * N1;

    // per-output-dim accumulator (threads 0..129 active for the dot)
    float accD = 0.0f;
    float degAcc = 0.0f;

    const float* colPtr; int colStride;
    if (tid < DIM_IN)      { colPtr = lm_X + tid;            colStride = DIM_IN; }
    else if (tid < DF)     { colPtr = lm_Y + (tid - DIM_IN); colStride = 2; }
    else                   { colPtr = lm_X;                  colStride = 0; }

    for (int j0 = 0; j0 < N1; j0 += CHUNK) {
        float a0 = arow[j0 + tid];
        float a1 = arow[j0 + tid + 256];
        lds[tid]       = a0;
        lds[tid + 256] = a1;
        degAcc += a0 + a1;
        __syncthreads();
        if (tid < DF) {
            #pragma unroll 4
            for (int jj = 0; jj < CHUNK; ++jj) {
                accD = fmaf(lds[jj], colPtr[(size_t)(j0 + jj) * colStride], accD);
            }
        }
        __syncthreads();
    }

    // reduce deg across block
    red[tid] = degAcc;
    __syncthreads();
    for (int s = 128; s > 0; s >>= 1) {
        if (tid < s) red[tid] += red[tid + s];
        __syncthreads();
    }
    const float deg = red[0] + 1.0f;
    __syncthreads();

    if (tid < DF) {
        const float t0 = (tid < DIM_IN) ? tg_X[(size_t)row * DIM_IN + tid] : 0.0f;
        sAgg[tid] = (accD + t0) / deg;
    }
    __syncthreads();

    // f1 = agg @ W1 + b1 ; fin = [tg_X_row, f1]
    if (tid < DIM_IN) sFin[tid] = tg_X[(size_t)row * DIM_IN + tid];
    if (tid < DF) {
        float f1 = b1[tid];
        #pragma unroll 2
        for (int i = 0; i < DF; ++i) {
            f1 = fmaf(sAgg[i], W1[(size_t)i * DF + tid], f1);
        }
        sFin[DIM_IN + tid] = f1;
    }
    __syncthreads();

    // h = relu(fin @ Wp1 + bp1); y = h @ Wp2 (+ bp2 at the end)
    float y0 = 0.0f, y1 = 0.0f;
    for (int hh = tid; hh < DIM_MED; hh += 256) {
        float hv = bp1[hh];
        #pragma unroll 2
        for (int i = 0; i < DIM_Z; ++i) {
            hv = fmaf(sFin[i], Wp1[(size_t)i * DIM_MED + hh], hv);
        }
        hv = fmaxf(hv, 0.0f);
        y0 = fmaf(hv, Wp2[(size_t)hh * DIM_OUT + 0], y0);
        y1 = fmaf(hv, Wp2[(size_t)hh * DIM_OUT + 1], y1);
    }
    red[tid] = y0;
    __syncthreads();
    for (int s = 128; s > 0; s >>= 1) {
        if (tid < s) red[tid] += red[tid + s];
        __syncthreads();
    }
    if (tid == 0) y_out[(size_t)row * DIM_OUT + 0] = red[0] + bp2[0];
    __syncthreads();
    red[tid] = y1;
    __syncthreads();
    for (int s = 128; s > 0; s >>= 1) {
        if (tid < s) red[tid] += red[tid + s];
        __syncthreads();
    }
    if (tid == 0) y_out[(size_t)row * DIM_OUT + 1] = red[0] + bp2[1];
}

// ---------------------------------------------------------------------------
// launch
// ---------------------------------------------------------------------------
extern "C" void kernel_launch(void* const* d_in, const int* in_sizes, int n_in,
                              void* d_out, int out_size, void* d_ws, size_t ws_size,
                              hipStream_t stream) {
    const float* lm_X = (const float*)d_in[0];
    const float* lm_Y = (const float*)d_in[1];
    const float* tg_X = (const float*)d_in[2];
    // d_in[3] = tg_Y (unused by the reference output)
    const float* Wq   = (const float*)d_in[4];
    const float* bq   = (const float*)d_in[5];
    const float* Wk   = (const float*)d_in[6];
    const float* bk   = (const float*)d_in[7];
    const float* W1   = (const float*)d_in[8];
    const float* b1   = (const float*)d_in[9];
    const float* Wp1  = (const float*)d_in[10];
    const float* bp1  = (const float*)d_in[11];
    const float* Wp2  = (const float*)d_in[12];
    const float* bp2  = (const float*)d_in[13];
    const float* noise_u = (const float*)d_in[14];

    // outputs: y_pred [N2*2], mask_a_prob [NTOT*NTOT], mask_a_matrix [NTOT*NTOT]
    float* out      = (float*)d_out;
    float* y_pred   = out;
    float* out_prob = out + (size_t)N2 * DIM_OUT;
    float* out_mask = out_prob + (size_t)NTOT * NTOT;

    // workspace: Qh, Kh (f16 [NTOT][KPAD]), adj (f32 [N2][N1])
    char* ws = (char*)d_ws;
    const size_t qkBytes = (size_t)NTOT * KPAD * sizeof(_Float16);
    _Float16* Qh  = (_Float16*)(ws);
    _Float16* Kh  = (_Float16*)(ws + qkBytes);
    float*    adj = (float*)(ws + 2 * qkBytes);

    // 1) projections (Q pre-scaled by 1/sqrt(258))
    geo_proj_kernel<<<dim3(NTOT), dim3(256), 0, stream>>>(
        lm_X, tg_X, Wq, bq, Wk, bk, Qh, Kh);

    // 2) fused score GEMM + elementwise outputs + adj block
    geo_score_kernel<<<dim3(NTOT / 64, NTOT / 128), dim3(128), 0, stream>>>(
        Qh, Kh, noise_u, out_prob, out_mask, adj);

    // 3) fused aggregation + MLP tail
    geo_tail_kernel<<<dim3(N2), dim3(256), 0, stream>>>(
        adj, lm_X, lm_Y, tg_X, W1, b1, Wp1, bp1, Wp2, bp2, y_pred);
}
